// GenNet_42082089566475
// MI455X (gfx1250) — matrix operand order repaired and checked by reading, not compiled, not run on hardware
//
#include <hip/hip_runtime.h>

// Problem constants (match reference)
#define BN  512   // batch
#define TN  512   // time steps
#define IND 10    // input dim
#define HN  128   // hidden dim
#define EN  300   // embed dim
// w_ih rows total = 2 layers * 4H = 1024; fc2_w rows = 300

typedef __bf16 bf16_t;
typedef __attribute__((ext_vector_type(16))) __bf16 v16bf;
typedef __attribute__((ext_vector_type(8)))  __bf16 v8bf;
typedef __attribute__((ext_vector_type(8)))  float  v8f;

union BFrag { v16bf v; v8bf h[2]; };

__device__ __forceinline__ bf16_t f2bf(float f) {
  union { float f; unsigned u; } a; a.f = f;
  unsigned r = a.u + 0x7FFFu + ((a.u >> 16) & 1u);   // round-to-nearest-even
  unsigned short hb = (unsigned short)(r >> 16);
  bf16_t o; __builtin_memcpy(&o, &hb, 2); return o;
}

// Fast activations: v_exp_f32 + v_rcp_f32, no IEEE division sequences.
// Saturation is handled naturally by exp/rcp infinities, no clamps needed.
__device__ __forceinline__ float sigf(float x) {
  return __builtin_amdgcn_rcpf(1.0f + __expf(-x));
}
__device__ __forceinline__ float tanh_fast(float x) {
  return 1.0f - 2.0f * __builtin_amdgcn_rcpf(1.0f + __expf(2.0f * x));
}

// A-fragment (16x32 bf16, MxK) from row-major [16][128] tile.
// ISA 7.12.2: lanes 0-15: V0-3 = K 0..7, V4-7 = K 16..23 ; lanes 16-31: +8.
__device__ __forceinline__ v16bf load_A_lds(const bf16_t* x, int lane, int kt) {
  const int row = lane & 15;
  const int hi  = lane >> 4;
  const bf16_t* p = x + row * HN + kt * 32 + hi * 8;
  BFrag f;
  f.h[0] = *(const v8bf*)(p);
  f.h[1] = *(const v8bf*)(p + 16);
  return f.v;
}

// B-fragment (32x16 bf16, KxN) where B[k][n] = W[row][k], W row-major [*,128].
// lanes 0-15 hold K=kt*32+{0..7,16..23} for col N=lane; lanes 16-31: +8.
__device__ __forceinline__ v16bf load_B_row(const bf16_t* W, int row, int hi, int kt) {
  const bf16_t* p = W + (size_t)row * HN + kt * 32 + hi * 8;
  BFrag f;
  f.h[0] = *(const v8bf*)(p);
  f.h[1] = *(const v8bf*)(p + 16);
  return f.v;
}

__device__ __forceinline__ v8f wmma_bf16(v16bf a, v16bf b, v8f c) {
  return __builtin_amdgcn_wmma_f32_16x16x32_bf16(
      /*neg_a=*/false, a, /*neg_b=*/false, b,
      /*c_mod=*/(short)0, c, /*reuse_a=*/false, /*reuse_b=*/false);
}

// ---------------- prep: fp32 -> bf16 weight conversion into d_ws ----------------
__global__ void prep_weights(const float* __restrict__ w_ih,
                             const float* __restrict__ fc2_w,
                             bf16_t* __restrict__ wih_b,
                             bf16_t* __restrict__ fc2_b16) {
  const int NW = 2 * 4 * HN * HN;   // 131072
  const int NF = EN * HN;           // 38400
  int i = blockIdx.x * 256 + threadIdx.x;
  if (i < NW) wih_b[i] = f2bf(w_ih[i]);
  int j = i - NW;
  if (j >= 0 && j < NF) fc2_b16[j] = f2bf(fc2_w[j]);
}

// ---------------- fused recurrence + fc2 ----------------
// Grid: 32 blocks (one per 16-row batch tile), 256 threads = 8 waves.
// Wave w owns hidden columns [16w, 16w+16) of the gate GEMMs and output
// column tiles {w, w+8, w+16} of fc2.
__global__ __launch_bounds__(256, 1)
void gen_net_kernel(const float* __restrict__ seq,     // [512,10]
                    const float* __restrict__ fc1_w,   // [128,10]
                    const float* __restrict__ fc1_b,   // [128]
                    const float* __restrict__ b_ih,    // [2,512] flat
                    const float* __restrict__ b_hh,    // [2,512] flat
                    const float* __restrict__ fc2_b,   // [300]
                    const bf16_t* __restrict__ WIH,    // [1024,128] bf16
                    const bf16_t* __restrict__ W2,     // [300,128] bf16
                    float* __restrict__ out)           // [512,512,300]
{
  __shared__ bf16_t buf0[16 * HN];   // x_t  (ping)
  __shared__ bf16_t buf1[16 * HN];   // h1   (pong)

  const int tid  = threadIdx.x;
  const int lane = tid & 31;
  const int wid  = tid >> 5;         // 0..7
  const int n    = lane & 15;        // output column within 16-wide tile
  const int hi   = lane >> 4;        // lane half
  const int row0 = blockIdx.x * 16;  // batch-row base

  // ---- fc1 prologue: x0 = seq @ fc1_wT + fc1_b -> buf0 (bf16) ----
  for (int e = tid; e < 16 * HN; e += 256) {
    int r = e >> 7, hc = e & 127;
    float s = fc1_b[hc];
#pragma unroll
    for (int k = 0; k < IND; ++k)
      s += seq[(row0 + r) * IND + k] * fc1_w[hc * IND + k];
    buf0[r * HN + hc] = f2bf(s);
  }

  // ---- hoist recurrent weights (i,g,o gates; f-gate is dead: c_prev==0) ----
  v16bf Wf[2][3][4];                 // [layer][gate][ktile] -> 192 VGPRs
  float bias[2][3];
#pragma unroll
  for (int l = 0; l < 2; ++l) {
#pragma unroll
    for (int g = 0; g < 3; ++g) {
      const int goff = (g == 0) ? 0 : (g == 1) ? 256 : 384;  // i, g, o
      const int j0 = l * 512 + goff + 16 * wid;
      bias[l][g] = b_ih[j0 + n] + b_hh[j0 + n];
#pragma unroll
      for (int kt = 0; kt < 4; ++kt)
        Wf[l][g][kt] = load_B_row(WIH, j0 + n, hi, kt);
    }
  }

  // fc2 per-lane column state (3 tiles per wave; tiles with col>=300 are masked)
  int   col[3], jr[3];
  float fbias[3];
#pragma unroll
  for (int q = 0; q < 3; ++q) {
    col[q]   = 16 * (wid + 8 * q) + n;
    jr[q]    = (col[q] < EN) ? col[q] : (EN - 1);   // clamped row for loads
    fbias[q] = fc2_b[jr[q]];
  }

  __syncthreads();
  v16bf xf[4];
#pragma unroll
  for (int kt = 0; kt < 4; ++kt) xf[kt] = load_A_lds(buf0, lane, kt);

  for (int t = 0; t < TN; ++t) {
    // ---------- layer 1: h1 = act(x_t @ W1T + b1) -> buf1 ----------
    {
      v8f acc[3];
#pragma unroll
      for (int g = 0; g < 3; ++g) {
        float bsc = bias[0][g];
        v8f c = { bsc, bsc, bsc, bsc, bsc, bsc, bsc, bsc };
#pragma unroll
        for (int kt = 0; kt < 4; ++kt)
          c = wmma_bf16(xf[kt], Wf[0][g][kt], c);
        acc[g] = c;
      }
#pragma unroll
      for (int v = 0; v < 8; ++v) {
        float ci = sigf(acc[0][v]) * tanh_fast(acc[1][v]);
        float h  = sigf(acc[2][v]) * tanh_fast(ci);
        buf1[(v + 8 * hi) * HN + 16 * wid + n] = f2bf(h);
      }
    }
    __syncthreads();
    v16bf hf[4];
#pragma unroll
    for (int kt = 0; kt < 4; ++kt) hf[kt] = load_A_lds(buf1, lane, kt);

    // ---------- layer 2: x_{t+1} = act(h1 @ W2T + b2) -> buf0 ----------
    {
      v8f acc[3];
#pragma unroll
      for (int g = 0; g < 3; ++g) {
        float bsc = bias[1][g];
        v8f c = { bsc, bsc, bsc, bsc, bsc, bsc, bsc, bsc };
#pragma unroll
        for (int kt = 0; kt < 4; ++kt)
          c = wmma_bf16(hf[kt], Wf[1][g][kt], c);
        acc[g] = c;
      }
#pragma unroll
      for (int v = 0; v < 8; ++v) {
        float ci = sigf(acc[0][v]) * tanh_fast(acc[1][v]);
        float h  = sigf(acc[2][v]) * tanh_fast(ci);
        buf0[(v + 8 * hi) * HN + 16 * wid + n] = f2bf(h);
      }
    }
    __syncthreads();
#pragma unroll
    for (int kt = 0; kt < 4; ++kt) xf[kt] = load_A_lds(buf0, lane, kt);

    // ---------- fused fc2: out[b, t, :] = x_{t+1} @ fc2_wT + fc2_b ----------
    // Reuses the x_{t+1} A-fragments just loaded for the next iteration.
#pragma unroll
    for (int q = 0; q < 3; ++q) {
      float bsc = fbias[q];
      v8f c = { bsc, bsc, bsc, bsc, bsc, bsc, bsc, bsc };
#pragma unroll
      for (int kt = 0; kt < 4; ++kt) {
        v16bf bf = load_B_row(W2, jr[q], hi, kt);   // L2-resident reload
        c = wmma_bf16(xf[kt], bf, c);
      }
      if (col[q] < EN) {
#pragma unroll
        for (int v = 0; v < 8; ++v) {
          int b = row0 + v + 8 * hi;
          out[((size_t)b * TN + t) * EN + col[q]] = c[v];
        }
      }
    }
  }
}

extern "C" void kernel_launch(void* const* d_in, const int* in_sizes, int n_in,
                              void* d_out, int out_size, void* d_ws, size_t ws_size,
                              hipStream_t stream) {
  const float* seq   = (const float*)d_in[0];   // [512,10]
  const float* fc1_w = (const float*)d_in[1];   // [128,10]
  const float* fc1_b = (const float*)d_in[2];   // [128]
  const float* w_ih  = (const float*)d_in[3];   // [2,512,128]
  // d_in[4] = w_hh: mathematically unused (h0 == 0 every step)
  const float* b_ih  = (const float*)d_in[5];   // [2,512]
  const float* b_hh  = (const float*)d_in[6];   // [2,512]
  const float* fc2_w = (const float*)d_in[7];   // [300,128]
  const float* fc2_b = (const float*)d_in[8];   // [300]
  float* out = (float*)d_out;

  // Workspace layout: bf16 w_ih (131072) | bf16 fc2_w (38400)  (~340 KB)
  bf16_t* wih_b = (bf16_t*)d_ws;
  bf16_t* fc2_b16 = wih_b + 2 * 4 * HN * HN;

  const int NTOT = 2 * 4 * HN * HN + EN * HN;
  prep_weights<<<(NTOT + 255) / 256, 256, 0, stream>>>(w_ih, fc2_w, wih_b, fc2_b16);

  gen_net_kernel<<<BN / 16, 256, 0, stream>>>(seq, fc1_w, fc1_b, b_ih, b_hh,
                                              fc2_b, wih_b, fc2_b16, out);
}
